// Model_62088047230979
// MI455X (gfx1250) — matrix operand order repaired and checked by reading, not compiled, not run on hardware
//
#include <hip/hip_runtime.h>
#include <hip/hip_bf16.h>
#include <math.h>

// ---------------------------------------------------------------------------
// PointNet++-MSG forward for gfx1250 (MI455X).
// All MLP layers run as bf16 WMMA GEMMs (v_wmma_f32_16x16x32_bf16) with f32
// accumulate. Each wave owns a 64x32 output macro-tile: 4 O-tiles x 2 M-tiles,
// so each K-step issues 8 WMMAs against 12 b128 loads (B fragments reused
// across 4 A fragments). The whole working set is L2-resident (192MB L2), so
// the GEMMs are issue-bound, not HBM-bound. BN is training-mode (two-pass:
// per-channel mean/var reduction, then normalize+ReLU+relayout to bf16).
// Irregular steps (FPS / ball-query / 3-NN interp / gathers) are compact
// kernels (FPS uses an LDS argmax reduction; the others are one thread per output).
// ---------------------------------------------------------------------------

typedef __bf16 bf16_t;
typedef bf16_t v16bf __attribute__((ext_vector_type(16)));
typedef bf16_t v8bf  __attribute__((ext_vector_type(8)));
typedef float  v8f   __attribute__((ext_vector_type(8)));

#define CDIV(a, b) (((a) + (b) - 1) / (b))
static inline int rup32(int x) { return (x + 31) & ~31; }
static inline int rup64(int x) { return (x + 63) & ~63; }

#define BATCH 2
#define NPTS  8192
#define KNB   5
#define FEATC 64

// ---------------------------------------------------------------------------
// Kernel: per-cloud normalize (center + scale to unit sphere). One block/batch.
// ---------------------------------------------------------------------------
__global__ void k_normalize(const float* __restrict__ pts, float* __restrict__ pc, int N) {
  int b = blockIdx.x;
  const float* p = pts + (size_t)b * N * 3;
  float* q = pc + (size_t)b * N * 3;
  __shared__ float r0[256], r1[256], r2s[256];
  __shared__ float mean[3];
  __shared__ float scale;
  int tid = threadIdx.x;
  float sx = 0.f, sy = 0.f, sz = 0.f;
  for (int i = tid; i < N; i += blockDim.x) {
    sx += p[3 * i]; sy += p[3 * i + 1]; sz += p[3 * i + 2];
  }
  r0[tid] = sx; r1[tid] = sy; r2s[tid] = sz;
  __syncthreads();
  for (int st = 128; st > 0; st >>= 1) {
    if (tid < st) { r0[tid] += r0[tid + st]; r1[tid] += r1[tid + st]; r2s[tid] += r2s[tid + st]; }
    __syncthreads();
  }
  if (tid == 0) { mean[0] = r0[0] / N; mean[1] = r1[0] / N; mean[2] = r2s[0] / N; }
  __syncthreads();
  float mx = 0.f;
  for (int i = tid; i < N; i += blockDim.x) {
    float dx = p[3 * i] - mean[0], dy = p[3 * i + 1] - mean[1], dz = p[3 * i + 2] - mean[2];
    float d = sqrtf(dx * dx + dy * dy + dz * dz);
    mx = fmaxf(mx, d);
  }
  r0[tid] = mx;
  __syncthreads();
  for (int st = 128; st > 0; st >>= 1) {
    if (tid < st) r0[tid] = fmaxf(r0[tid], r0[tid + st]);
    __syncthreads();
  }
  if (tid == 0) scale = r0[0];
  __syncthreads();
  float inv = 1.f / scale;
  for (int i = tid; i < N; i += blockDim.x) {
    q[3 * i]     = (p[3 * i]     - mean[0]) * inv;
    q[3 * i + 1] = (p[3 * i + 1] - mean[1]) * inv;
    q[3 * i + 2] = (p[3 * i + 2] - mean[2]) * inv;
  }
}

// ---------------------------------------------------------------------------
// Kernel: farthest point sampling. One block per batch; S sequential rounds of
// distance-update + LDS argmax reduction (this part is inherently serial).
// ---------------------------------------------------------------------------
__global__ void k_fps(const float* __restrict__ xyz, int n, int S,
                      float* __restrict__ dist, int* __restrict__ out) {
  int b = blockIdx.x;
  const float* x = xyz + (size_t)b * n * 3;
  float* dd = dist + (size_t)b * n;
  int* o = out + (size_t)b * S;
  __shared__ float rmax[256];
  __shared__ int rarg[256];
  __shared__ float cen[3];
  __shared__ int sfar;
  int tid = threadIdx.x;
  for (int i = tid; i < n; i += blockDim.x) dd[i] = 1e10f;
  if (tid == 0) sfar = 0;
  __syncthreads();
  for (int it = 0; it < S; ++it) {
    if (tid == 0) {
      int f = sfar;
      o[it] = f;
      cen[0] = x[3 * f]; cen[1] = x[3 * f + 1]; cen[2] = x[3 * f + 2];
    }
    __syncthreads();
    float bm = -1.f; int bi = 0;
    for (int i = tid; i < n; i += blockDim.x) {
      float dx = x[3 * i] - cen[0], dy = x[3 * i + 1] - cen[1], dz = x[3 * i + 2] - cen[2];
      float d = dx * dx + dy * dy + dz * dz;
      float nv = fminf(dd[i], d);
      dd[i] = nv;
      if (nv > bm) { bm = nv; bi = i; }
    }
    rmax[tid] = bm; rarg[tid] = bi;
    __syncthreads();
    for (int st = 128; st > 0; st >>= 1) {
      if (tid < st && rmax[tid + st] > rmax[tid]) { rmax[tid] = rmax[tid + st]; rarg[tid] = rarg[tid + st]; }
      __syncthreads();
    }
    if (tid == 0) sfar = rarg[0];
    __syncthreads();
  }
}

// ---------------------------------------------------------------------------
// Kernel: gather sampled centroids' coordinates.
// ---------------------------------------------------------------------------
__global__ void k_gather_xyz(const float* __restrict__ xyz, const int* __restrict__ idx,
                             float* __restrict__ nxyz, int n, int S, int total) {
  int t = blockIdx.x * blockDim.x + threadIdx.x;
  if (t >= total) return;
  int b = t / S;
  int j = idx[t];
  const float* src = xyz + ((size_t)b * n + j) * 3;
  float* dst = nxyz + (size_t)t * 3;
  dst[0] = src[0]; dst[1] = src[1]; dst[2] = src[2];
}

// ---------------------------------------------------------------------------
// Kernel: ball query. Equivalent to reference's sort(where(d>r^2,n,arange)):
// collect the first ns in-radius indices in increasing index order, pad with
// the first hit (self is always in-radius so cnt >= 1).
// ---------------------------------------------------------------------------
__global__ void k_qball(const float* __restrict__ xyz, const float* __restrict__ nxyz,
                        float r2, int ns, int n, int S, int* __restrict__ gidx, int total) {
  int t = blockIdx.x * blockDim.x + threadIdx.x;
  if (t >= total) return;
  int b = t / S;
  const float* c = nxyz + (size_t)t * 3;
  const float* x = xyz + (size_t)b * n * 3;
  int* g = gidx + (size_t)t * ns;
  int cnt = 0, first = 0;
  for (int j = 0; j < n && cnt < ns; ++j) {
    float dx = x[3 * j] - c[0], dy = x[3 * j + 1] - c[1], dz = x[3 * j + 2] - c[2];
    float d = dx * dx + dy * dy + dz * dz;
    if (d <= r2) {
      if (cnt == 0) first = j;
      g[cnt++] = j;
    }
  }
  if (cnt == 0) { g[0] = 0; cnt = 1; first = 0; }
  for (; cnt < ns; ++cnt) g[cnt] = first;
}

// ---------------------------------------------------------------------------
// Kernel: build grouped GEMM input X[M][Cpad] (bf16, K-contiguous rows):
// channels = [prev feats (Cp), relative xyz (3), zero pad].
// m = (b*S + s)*ns + j  (ns innermost so max-pool is over contiguous rows).
// ---------------------------------------------------------------------------
__global__ void k_group(const float* __restrict__ xyz, const float* __restrict__ pts,
                        int Cp, int n, const float* __restrict__ nxyz,
                        const int* __restrict__ gidx, bf16_t* __restrict__ X,
                        int S, int ns, int Cpad, long long total) {
  long long t = (long long)blockIdx.x * blockDim.x + threadIdx.x;
  if (t >= total) return;
  int j = (int)(t % ns);
  long long u = t / ns;
  int s = (int)(u % S);
  int b = (int)(u / S);
  int g = gidx[((size_t)b * S + s) * ns + j];
  bf16_t* xr = X + (size_t)t * Cpad;
  int c = 0;
  for (; c < Cp; ++c) xr[c] = (bf16_t)pts[((size_t)b * Cp + c) * n + g];
  const float* gp = xyz + ((size_t)b * n + g) * 3;
  const float* cp = nxyz + ((size_t)b * S + s) * 3;
  xr[c]     = (bf16_t)(gp[0] - cp[0]);
  xr[c + 1] = (bf16_t)(gp[1] - cp[1]);
  xr[c + 2] = (bf16_t)(gp[2] - cp[2]);
  for (c += 3; c < Cpad; ++c) xr[c] = (bf16_t)0.f;
}

// ---------------------------------------------------------------------------
// Kernel: convert + pad weights f32[O][C] -> bf16[Opad64][Cpad32] (zeros pad).
// ---------------------------------------------------------------------------
__global__ void k_convw(const float* __restrict__ W, bf16_t* __restrict__ Wb,
                        int O, int C, int Cpad, long long total) {
  long long t = (long long)blockIdx.x * blockDim.x + threadIdx.x;
  if (t >= total) return;
  int c = (int)(t % Cpad);
  int o = (int)(t / Cpad);
  Wb[t] = (o < O && c < C) ? (bf16_t)W[(size_t)o * C + c] : (bf16_t)0.f;
}

// ---------------------------------------------------------------------------
// Kernel: WMMA GEMM  Z[o][m] = sum_c W[o][c] * X[m][c] + bias[o].
// One wave per 64x32 output macro-tile: 4 O-tiles x 2 M-tiles, K blocked by 32.
// Per K-step: 8 v_wmma_f32_16x16x32_bf16 vs 12 b128 loads (B reused 4x).
// A fragment (16x32): lanes 0-15 row M=lane, K {0..15};
//                     lanes 16-31 row M=lane-16, K {16..31}.
// B fragment (32x16): lane column N=lane&15, K = (lane<16?0:16)+e.
// D: VGPR r -> row o = tile_o + r + (lane>=16 ? 8 : 0), col m = tile_m + (lane&15).
// ---------------------------------------------------------------------------
__global__ __launch_bounds__(256)
void k_wmma_gemm(const bf16_t* __restrict__ W, const bf16_t* __restrict__ X,
                 const float* __restrict__ bias, float* __restrict__ Z,
                 int O, int Cpad, int Mt, int tiles, int tilesO) {
  int wave = (blockIdx.x * blockDim.x + threadIdx.x) >> 5;
  int lane = threadIdx.x & 31;
  if (wave >= tiles) return;  // wave-uniform: EXEC stays all-1s for WMMA
  int to = (wave % tilesO) * 64;
  int tm = (wave / tilesO) * 32;
  int row = lane & 15;
  int ka = (lane < 16) ? 0 : 16;
  int kb = (lane < 16) ? 0 : 16;
  const bf16_t* wbase = W + (size_t)(to + row) * Cpad;
  const bf16_t* x0 = X + (size_t)(tm + row) * Cpad;
  const bf16_t* x1 = X + (size_t)(tm + 16 + row) * Cpad;
  v8f acc[8] = {};  // acc[t*2 + mt] : O-tile t, M-tile mt
  for (int kk = 0; kk < Cpad; kk += 32) {
    __builtin_prefetch(x0 + kk + 128, 0, 1);  // -> global_prefetch_b8
    v16bf b0 = *(const v16bf*)(x0 + kk + kb);
    v16bf b1 = *(const v16bf*)(x1 + kk + kb);
#pragma unroll
    for (int t = 0; t < 4; ++t) {
      const bf16_t* wr = wbase + (size_t)t * 16 * Cpad;
      v8bf alo = *(const v8bf*)(wr + kk + ka);
      v8bf ahi = *(const v8bf*)(wr + kk + 8 + ka);
      v16bf a;
#pragma unroll
      for (int i = 0; i < 8; ++i) { a[i] = alo[i]; a[i + 8] = ahi[i]; }
      acc[t * 2] = __builtin_amdgcn_wmma_f32_16x16x32_bf16(
          false, a, false, b0, (short)0, acc[t * 2], false, false);
      acc[t * 2 + 1] = __builtin_amdgcn_wmma_f32_16x16x32_bf16(
          false, a, false, b1, (short)0, acc[t * 2 + 1], false, false);
    }
  }
  int obase = to + ((lane < 16) ? 0 : 8);
#pragma unroll
  for (int t = 0; t < 4; ++t) {
#pragma unroll
    for (int mt = 0; mt < 2; ++mt) {
      int m = tm + mt * 16 + row;
      v8f c = acc[t * 2 + mt];
#pragma unroll
      for (int r = 0; r < 8; ++r) {
        int o = obase + t * 16 + r;
        if (o < O) Z[(size_t)o * Mt + m] = c[r] + bias[o];
      }
    }
  }
}

// ---------------------------------------------------------------------------
// Kernel: per-channel BN stats (biased var) over all M. One block per channel.
// ---------------------------------------------------------------------------
__global__ void k_bn_stats(const float* __restrict__ Z, int Mt,
                           float* __restrict__ mean, float* __restrict__ var) {
  int o = blockIdx.x;
  const float* z = Z + (size_t)o * Mt;
  __shared__ float rs[256], rq[256];
  int tid = threadIdx.x;
  float s = 0.f, q = 0.f;
  for (int i = tid; i < Mt; i += blockDim.x) { float v = z[i]; s += v; q += v * v; }
  rs[tid] = s; rq[tid] = q;
  __syncthreads();
  for (int st = 128; st > 0; st >>= 1) {
    if (tid < st) { rs[tid] += rs[tid + st]; rq[tid] += rq[tid + st]; }
    __syncthreads();
  }
  if (tid == 0) {
    float m = rs[0] / Mt;
    mean[o] = m;
    var[o] = rq[0] / Mt - m * m;
  }
}

// ---------------------------------------------------------------------------
// Kernel: BN apply + ReLU; relayout to bf16 [M][Opad] for the next GEMM and/or
// write f32 channel-major [B][O][npts] (FP outputs).
// ---------------------------------------------------------------------------
__global__ void k_bn_apply(const float* __restrict__ Z, const float* __restrict__ mean,
                           const float* __restrict__ var, const float* __restrict__ g,
                           const float* __restrict__ be, bf16_t* __restrict__ Xn,
                           float* __restrict__ Fout, int O, int Opad, int Mt,
                           int npts, long long total) {
  long long t = (long long)blockIdx.x * blockDim.x + threadIdx.x;
  if (t >= total) return;
  int o = (int)(t % Opad);
  long long m = t / Opad;
  float y = 0.f;
  if (o < O) {
    float z = Z[(size_t)o * Mt + m];
    y = (z - mean[o]) * rsqrtf(var[o] + 1e-5f) * g[o] + be[o];
    y = fmaxf(y, 0.f);
  }
  if (Xn) Xn[(size_t)m * Opad + o] = (bf16_t)y;
  if (Fout && o < O) {
    int b = (int)(m / npts);
    int i = (int)(m % npts);
    Fout[((size_t)b * O + o) * npts + i] = y;
  }
}

// ---------------------------------------------------------------------------
// Kernel: max-pool over the ns axis -> pts f32 [B][Ctot][S] at channel offset.
// ---------------------------------------------------------------------------
__global__ void k_pool(const bf16_t* __restrict__ Xl, int Opad, int O, int S, int ns,
                       float* __restrict__ pts, int Ctot, int choff, int total) {
  int t = blockIdx.x * blockDim.x + threadIdx.x;
  if (t >= total) return;
  int s = t % S;
  int u = t / S;
  int o = u % O;
  int b = u / O;
  size_t m0 = ((size_t)b * S + s) * ns;
  float mx = 0.f;  // post-ReLU values are >= 0
  for (int j = 0; j < ns; ++j) mx = fmaxf(mx, (float)Xl[(m0 + j) * Opad + o]);
  pts[((size_t)b * Ctot + choff + o) * S + s] = mx;
}

// ---------------------------------------------------------------------------
// Kernel: 3-NN inverse-distance interpolation + concat -> X[M][Cpad] bf16.
// channels = [pts1 (D1), interp(pts2) (D2), zero pad].
// ---------------------------------------------------------------------------
__global__ void k_interp(const float* __restrict__ x1, const float* __restrict__ x2,
                         const float* __restrict__ pts1, int D1,
                         const float* __restrict__ pts2, int D2,
                         int n1, int s2, bf16_t* __restrict__ X, int Cpad, int total) {
  int t = blockIdx.x * blockDim.x + threadIdx.x;
  if (t >= total) return;
  int i = t % n1;
  int b = t / n1;
  const float* p = x1 + ((size_t)b * n1 + i) * 3;
  float d0 = 1e30f, d1 = 1e30f, d2v = 1e30f;
  int i0 = 0, i1 = 0, i2 = 0;
  const float* q = x2 + (size_t)b * s2 * 3;
  for (int s = 0; s < s2; ++s) {
    float dx = p[0] - q[3 * s], dy = p[1] - q[3 * s + 1], dz = p[2] - q[3 * s + 2];
    float d = dx * dx + dy * dy + dz * dz;
    if (d < d0)      { d2v = d1; i2 = i1; d1 = d0; i1 = i0; d0 = d; i0 = s; }
    else if (d < d1) { d2v = d1; i2 = i1; d1 = d; i1 = s; }
    else if (d < d2v){ d2v = d; i2 = s; }
  }
  float w0 = 1.f / (d0 + 1e-8f), w1 = 1.f / (d1 + 1e-8f), w2 = 1.f / (d2v + 1e-8f);
  float ws = 1.f / (w0 + w1 + w2);
  w0 *= ws; w1 *= ws; w2 *= ws;
  bf16_t* xr = X + (size_t)t * Cpad;
  int c = 0;
  for (; c < D1; ++c) xr[c] = (bf16_t)pts1[((size_t)b * D1 + c) * n1 + i];
  for (int d = 0; d < D2; ++d) {
    const float* pd = pts2 + ((size_t)b * D2 + d) * s2;
    xr[D1 + d] = (bf16_t)(w0 * pd[i0] + w1 * pd[i1] + w2 * pd[i2]);
  }
  for (c = D1 + D2; c < Cpad; ++c) xr[c] = (bf16_t)0.f;
}

// ---------------------------------------------------------------------------
// Kernel: decoder neighborhood gather: X[m][j*F+c] = p0[b][c][nbd[b][i][j]].
// ---------------------------------------------------------------------------
__global__ void k_gather_nbd(const float* __restrict__ p0, const int* __restrict__ nbd,
                             bf16_t* __restrict__ X, int N, int F, int K1,
                             int Cpad, int total) {
  int t = blockIdx.x * blockDim.x + threadIdx.x;
  if (t >= total) return;
  int i = t % N;
  int b = t / N;
  bf16_t* xr = X + (size_t)t * Cpad;
  for (int j = 0; j < K1; ++j) {
    int g = nbd[((size_t)b * N + i) * K1 + j];
    const float* src = p0 + (size_t)b * F * N + g;
    for (int c = 0; c < F; ++c) xr[j * F + c] = (bf16_t)src[(size_t)c * N];
  }
  for (int c = K1 * F; c < Cpad; ++c) xr[c] = (bf16_t)0.f;
}

// ---------------------------------------------------------------------------
// Kernel: dec2 (3x128) + sigmoid + spherical-coordinate normal output.
// out = [tpn (B*N*3 f32), norm (B*N*3 f32)].
// ---------------------------------------------------------------------------
__global__ void k_dec2(const bf16_t* __restrict__ X, const float* __restrict__ W2,
                       const float* __restrict__ b2, float* __restrict__ out,
                       int N, int Mt) {
  int t = blockIdx.x * blockDim.x + threadIdx.x;
  if (t >= Mt) return;
  const bf16_t* xr = X + (size_t)t * 128;
  float a[3];
#pragma unroll
  for (int r = 0; r < 3; ++r) {
    float s = b2[r];
    const float* wr = W2 + r * 128;
    for (int c = 0; c < 128; ++c) s += (float)xr[c] * wr[c];
    a[r] = 1.f / (1.f + expf(-s));
  }
  float* tp = out + (size_t)t * 3;
  tp[0] = a[0]; tp[1] = a[1]; tp[2] = a[2];
  const float twopi = 6.283185307179586f;
  float th = a[0] * twopi, ph = a[1] * twopi;
  float* nr = out + (size_t)Mt * 3 + (size_t)t * 3;
  nr[0] = cosf(th);
  nr[1] = sinf(th) * sinf(ph);
  nr[2] = sinf(th) * cosf(ph);
}

// ---------------------------------------------------------------------------
// Host orchestration
// ---------------------------------------------------------------------------
extern "C" void kernel_launch(void* const* d_in, const int* in_sizes, int n_in,
                              void* d_out, int out_size, void* d_ws, size_t ws_size,
                              hipStream_t stream) {
  (void)in_sizes; (void)n_in; (void)out_size; (void)ws_size;

  const float* points = (const float*)d_in[0];   // [B,N,3]
  const int* nbd      = (const int*)d_in[1];     // [B,N,K+1]
  int pi = 2;
  auto nxt = [&]() -> const float* { return (const float*)d_in[pi++]; };

  // ---- workspace bump allocator ----
  size_t off = 0;
  auto alloc = [&](size_t bytes) -> void* {
    void* p = (char*)d_ws + off;
    off += (bytes + 255) & ~(size_t)255;
    return p;
  };

  static const int SA_NP[4] = {1024, 256, 64, 16};
  static const float SA_R[4][2] = {{0.05f, 0.1f}, {0.1f, 0.2f}, {0.2f, 0.4f}, {0.4f, 0.8f}};
  static const int SA_NS[2] = {16, 32};
  static const int SA_CH[4][2][3] = {
      {{16, 16, 32}, {32, 32, 64}},
      {{64, 64, 128}, {64, 96, 128}},
      {{128, 196, 256}, {128, 196, 256}},
      {{256, 256, 512}, {256, 384, 512}}};
  static const int SA_CTOT[4] = {96, 256, 512, 1024};

  float* pc = (float*)alloc((size_t)BATCH * NPTS * 3 * 4);
  float* dist = (float*)alloc((size_t)BATCH * NPTS * 4);
  int* fpsidx = (int*)alloc((size_t)BATCH * 1024 * 4);
  int* gidx = (int*)alloc((size_t)BATCH * 1024 * 32 * 4);
  float* nxyz[4];
  float* ptsArr[4];
  for (int l = 0; l < 4; ++l) {
    nxyz[l] = (float*)alloc((size_t)BATCH * SA_NP[l] * 3 * 4);
    ptsArr[l] = (float*)alloc((size_t)BATCH * SA_CTOT[l] * SA_NP[l] * 4);
  }
  float* p3 = (float*)alloc((size_t)BATCH * 256 * 64 * 4);
  float* p2 = (float*)alloc((size_t)BATCH * 256 * 256 * 4);
  float* p1 = (float*)alloc((size_t)BATCH * 128 * 1024 * 4);
  float* p0 = (float*)alloc((size_t)BATCH * FEATC * NPTS * 4);
  bf16_t* Xa = (bf16_t*)alloc((size_t)13 * 1024 * 1024);
  bf16_t* Xb = (bf16_t*)alloc((size_t)13 * 1024 * 1024);
  bf16_t* Wb = (bf16_t*)alloc((size_t)2 * 1024 * 1024);
  float* Zb = (float*)alloc((size_t)17 * 1024 * 1024);
  float* meanb = (float*)alloc(1024 * 4);
  float* varb = (float*)alloc(1024 * 4);

  // ---- one fused GEMM+BN+ReLU layer ----
  auto run_layer = [&](const float* W, const float* bias, const float* g, const float* be,
                       int O, int C, int Mt, const bf16_t* Xin, bf16_t* Xout,
                       float* Fout, int npts) {
    int Cpad = rup32(C), Ow = rup64(O), Ox = rup32(O);
    {
      long long tot = (long long)Ow * Cpad;
      k_convw<<<(unsigned)CDIV(tot, 256), 256, 0, stream>>>(W, Wb, O, C, Cpad, tot);
    }
    {
      int tilesO = Ow / 64, tilesM = Mt / 32;
      int tiles = tilesO * tilesM;
      k_wmma_gemm<<<(unsigned)CDIV(tiles * 32, 256), 256, 0, stream>>>(
          Wb, Xin, bias, Zb, O, Cpad, Mt, tiles, tilesO);
    }
    k_bn_stats<<<(unsigned)O, 256, 0, stream>>>(Zb, Mt, meanb, varb);
    {
      long long tot = (long long)Ox * Mt;
      k_bn_apply<<<(unsigned)CDIV(tot, 256), 256, 0, stream>>>(
          Zb, meanb, varb, g, be, Xout, Fout, O, Ox, Mt, npts, tot);
    }
  };

  // ---- normalize ----
  k_normalize<<<BATCH, 256, 0, stream>>>(points, pc, NPTS);

  // ---- set-abstraction (MSG) levels ----
  const float* xyzp = pc;
  int np = NPTS;
  const float* ptsp = nullptr;
  int Cp = 0;
  for (int l = 0; l < 4; ++l) {
    int S = SA_NP[l];
    k_fps<<<BATCH, 256, 0, stream>>>(xyzp, np, S, dist, fpsidx);
    k_gather_xyz<<<(unsigned)CDIV(BATCH * S, 256), 256, 0, stream>>>(
        xyzp, fpsidx, nxyz[l], np, S, BATCH * S);
    int Ctot = SA_CTOT[l];
    int choff = 0;
    for (int br = 0; br < 2; ++br) {
      int ns = SA_NS[br];
      float r = SA_R[l][br];
      k_qball<<<(unsigned)CDIV(BATCH * S, 128), 128, 0, stream>>>(
          xyzp, nxyz[l], r * r, ns, np, S, gidx, BATCH * S);
      int Cin = Cp + 3;
      int Cpad = rup32(Cin);
      long long Mt = (long long)BATCH * S * ns;
      k_group<<<(unsigned)CDIV(Mt, 128), 128, 0, stream>>>(
          xyzp, ptsp, Cp, np, nxyz[l], gidx, Xa, S, ns, Cpad, Mt);
      bf16_t* cur = Xa;
      bf16_t* nxb = Xb;
      int C = Cin, O = 0;
      for (int li = 0; li < 3; ++li) {
        const float* w = nxt();
        const float* bb = nxt();
        const float* gg = nxt();
        const float* bee = nxt();
        O = SA_CH[l][br][li];
        run_layer(w, bb, gg, bee, O, C, (int)Mt, cur, nxb, nullptr, 1);
        bf16_t* t = cur; cur = nxb; nxb = t;
        C = O;
      }
      k_pool<<<(unsigned)CDIV(BATCH * O * S, 256), 256, 0, stream>>>(
          cur, rup32(O), O, S, ns, ptsArr[l], Ctot, choff, BATCH * O * S);
      choff += O;
    }
    xyzp = nxyz[l];
    np = S;
    ptsp = ptsArr[l];
    Cp = Ctot;
  }

  // ---- feature propagation levels ----
  auto run_fp = [&](const float* x1, int n1, const float* x2, int s2,
                    const float* pts1, int D1, const float* pts2, int D2,
                    const int* chans, int nl, float* outp) {
    int Cin = D1 + D2;
    int Cpad = rup32(Cin);
    int Mt = BATCH * n1;
    k_interp<<<(unsigned)CDIV(Mt, 128), 128, 0, stream>>>(
        x1, x2, pts1, D1, pts2, D2, n1, s2, Xa, Cpad, Mt);
    bf16_t* cur = Xa;
    bf16_t* nxb = Xb;
    int C = Cin;
    for (int li = 0; li < nl; ++li) {
      const float* w = nxt();
      const float* bb = nxt();
      const float* gg = nxt();
      const float* bee = nxt();
      int O = chans[li];
      bool last = (li == nl - 1);
      run_layer(w, bb, gg, bee, O, C, Mt, cur, last ? nullptr : nxb,
                last ? outp : nullptr, n1);
      bf16_t* t = cur; cur = nxb; nxb = t;
      C = O;
    }
  };
  static const int fp0c[2] = {256, 256};
  static const int fp1c[2] = {256, 256};
  static const int fp2c[2] = {256, 128};
  static const int fp3c[4] = {128, 128, 128, FEATC};
  run_fp(nxyz[2], 64,   nxyz[3], 16,  ptsArr[2], 512, ptsArr[3], 1024, fp0c, 2, p3);
  run_fp(nxyz[1], 256,  nxyz[2], 64,  ptsArr[1], 256, p3, 256,        fp1c, 2, p2);
  run_fp(nxyz[0], 1024, nxyz[1], 256, ptsArr[0], 96,  p2, 256,        fp2c, 2, p1);
  run_fp(pc,      NPTS, nxyz[0], 1024, nullptr,  0,   p1, 128,        fp3c, 4, p0);

  // ---- decoder ----
  {
    int Cpad = rup32((KNB + 1) * FEATC);  // 384
    int Mt = BATCH * NPTS;
    k_gather_nbd<<<(unsigned)CDIV(Mt, 128), 128, 0, stream>>>(
        p0, nbd, Xa, NPTS, FEATC, KNB + 1, Cpad, Mt);
    const float* w1 = nxt();
    const float* b1 = nxt();
    const float* g1 = nxt();
    const float* be1 = nxt();
    run_layer(w1, b1, g1, be1, 128, (KNB + 1) * FEATC, Mt, Xa, Xb, nullptr, 1);
    const float* w2 = nxt();
    const float* b2 = nxt();
    nxt(); nxt();  // dec2 gamma/beta unused by reference
    k_dec2<<<(unsigned)CDIV(Mt, 256), 256, 0, stream>>>(Xb, w2, b2, (float*)d_out, NPTS, Mt);
  }
}
